// GAT_52819507806390
// MI455X (gfx1250) — matrix operand order repaired and checked by reading, not compiled, not run on hardware
//
#include <hip/hip_runtime.h>
#include <hip/hip_bf16.h>
#include <math.h>

// Problem constants (from reference)
#define Nn           50000
#define Ee           1600000
#define F_IN_        256
#define Hh           4
#define Cc           32
#define HC_          128
#define NUM_CLASSES_ 16
#define NUM_GRAPHS_  64
#define NEG_SLOPE_   0.2f
#define ETOT         (Ee + Nn)     // 1,650,000 (divisible by 8)
#define MTILES       (Nn / 16)     // 3125 exact

typedef __attribute__((ext_vector_type(16))) __bf16 v16bf;
typedef __attribute__((ext_vector_type(8)))  __bf16 v8bf;
typedef __attribute__((ext_vector_type(8)))  float  v8f;
typedef __attribute__((ext_vector_type(4)))  unsigned int v4u;
typedef __attribute__((ext_vector_type(8)))  int v8i;
typedef __attribute__((ext_vector_type(4)))  int v4i;

#if defined(__has_builtin)
#if __has_builtin(__builtin_amdgcn_tensor_load_to_lds) && __has_builtin(__builtin_amdgcn_s_wait_tensorcnt)
#define HAVE_TDM 1
#endif
#endif

// ---------- helpers ----------
__device__ __forceinline__ __bf16 f2bf(float f) {
    union { float f; unsigned u; } uf; uf.f = f;
    unsigned u = uf.u;
    unsigned r = (u + 0x7FFFu + ((u >> 16) & 1u)) >> 16;   // RNE
    union { unsigned short s; __bf16 b; } ub; ub.s = (unsigned short)r;
    return ub.b;
}
__device__ __forceinline__ unsigned bfpack2(float a, float b) {
    unsigned ua = __float_as_uint(a); ua = (ua + 0x7FFFu + ((ua >> 16) & 1u)) >> 16;
    unsigned ub = __float_as_uint(b); ub = (ub + 0x7FFFu + ((ub >> 16) & 1u)) >> 16;
    return (ua & 0xFFFFu) | (ub << 16);
}
// monotone encoding of float into unsigned so atomicMax(uint) == float max
__device__ __forceinline__ unsigned fenc(float f) {
    unsigned u = __float_as_uint(f);
    return (u & 0x80000000u) ? ~u : (u | 0x80000000u);
}
__device__ __forceinline__ float fdec(unsigned u) {
    unsigned b = (u & 0x80000000u) ? (u & 0x7FFFFFFFu) : ~u;
    return __uint_as_float(b);
}

// ---------- zero fill (float4) ----------
__global__ void zero4_kernel(float4* __restrict__ p, long long count4) {
    long long t = (long long)blockIdx.x * blockDim.x + threadIdx.x;
    if (t < count4) p[t] = make_float4(0.f, 0.f, 0.f, 0.f);
}

// ---------- fp32 -> bf16 streaming convert (4 elems/thread) ----------
__global__ void cvt_bf16_kernel(const float4* __restrict__ in, uint2* __restrict__ outp,
                                long long count4) {
    long long t = (long long)blockIdx.x * blockDim.x + threadIdx.x;
    if (t >= count4) return;
    float4 v = in[t];
    uint2 o; o.x = bfpack2(v.x, v.y); o.y = bfpack2(v.z, v.w);
    outp[t] = o;
}

// ---------- W [K x 128] fp32 -> Wt [128 x K] bf16 (transpose) ----------
__global__ void wtrans_kernel(const float* __restrict__ W, __bf16* __restrict__ Wt, int K) {
    int t = blockIdx.x * blockDim.x + threadIdx.x;   // over K*128
    if (t >= K * HC_) return;
    int k = t >> 7, n = t & 127;
    Wt[(size_t)n * K + k] = f2bf(W[t]);
}

// ---------- dense GEMM: D[N x 128] = A[N x K](bf16) * Wt[128 x K](bf16), WMMA ----------
// Wt staged into LDS once per workgroup via the Tensor Data Mover (TDM);
// one wave per 16-row tile, all 8 column tiles; EXEC all-ones per wave.
__global__ __launch_bounds__(128) void gemm_wmma_bf16(
    const __bf16* __restrict__ A, const __bf16* __restrict__ Wt,
    float* __restrict__ D, int K)
{
    __shared__ __align__(32) __bf16 ldsW[HC_ * F_IN_];   // 64 KB max (K <= 256)

    const int lane = threadIdx.x & 31;
    const int wave = threadIdx.x >> 5;
    const unsigned nelem = (unsigned)(HC_ * K);          // <= 32768 elements (bf16)

#if defined(HAVE_TDM)
    if (wave == 0) {                                     // wave-uniform; EXEC ignored by TDM
        unsigned long long ga = (unsigned long long)(uintptr_t)Wt;
        unsigned ldsoff = (unsigned)(uintptr_t)ldsW;     // LDS byte offset (addr[31:0])
        // D# group 0: count=1 | lds_addr | global_addr[56:0] | type=2
        v4u g0;
        g0[0] = 1u;
        g0[1] = ldsoff;
        g0[2] = (unsigned)(ga & 0xFFFFFFFFu);
        g0[3] = (unsigned)((ga >> 32) & 0x01FFFFFFu) | (2u << 30);
        // D# group 1: 1-D tile of nelem 2-byte elements
        v8i g1;
        g1[0] = (int)(1u << 16);                         // data_size=1 (2 bytes), mask=0
        g1[1] = (int)((nelem & 0xFFFFu) << 16);          // tensor_dim0[15:0]  (bits 63:48)
        g1[2] = (int)(((nelem >> 16) & 0xFFFFu)          // tensor_dim0[31:16]
                      | (1u << 16));                     // tensor_dim1 = 1
        g1[3] = (int)(nelem << 16);                      // tile_dim0 (bits 127:112)
        g1[4] = 0;                                       // tile_dim1=0, tile_dim2=0
        g1[5] = (int)nelem;                              // tensor_dim0_stride[31:0]
        g1[6] = 0; g1[7] = 0;
        v4i gz4 = {0, 0, 0, 0};
        v8i gz8 = {0, 0, 0, 0, 0, 0, 0, 0};
        __builtin_amdgcn_tensor_load_to_lds(g0, g1, gz4, gz4, gz8, 0);
        __builtin_amdgcn_s_wait_tensorcnt(0);
    }
#else
    for (unsigned i = threadIdx.x; i < nelem / 8; i += 128)   // fallback: b128 cooperative copy
        ((uint4*)ldsW)[i] = ((const uint4*)Wt)[i];
#endif
    __syncthreads();

    const int mtile = blockIdx.x * 4 + wave;
    if (mtile >= MTILES) return;           // wave-uniform guard (after barrier)
    const int m0   = mtile << 4;
    const int half = lane >> 4;            // 0 or 1
    const int l15  = lane & 15;

    const __bf16* arow = A + (size_t)(m0 + l15) * K;   // this lane's A row

    v8f acc[8] = {};

    for (int kc = 0; kc < K; kc += 32) {
        if (kc + 32 < K) __builtin_prefetch(arow + kc + 32, 0, 0);
        // A fragment (ISA 7.12.2 16-bit A 16x32): elements [0..7]  = K kc+half*8 .. +7
        //                                         elements [8..15] = K kc+16+half*8 .. +7
        v8bf lo = *(const v8bf*)(arow + kc + (half << 3));
        v8bf hi = *(const v8bf*)(arow + kc + 16 + (half << 3));
        v16bf a = __builtin_shufflevector(lo, hi, 0, 1, 2, 3, 4, 5, 6, 7,
                                          8, 9, 10, 11, 12, 13, 14, 15);
        #pragma unroll
        for (int t = 0; t < 8; ++t) {
            // B fragment from LDS: column n = t*16+l15, K = kc+half*8 .. +7, then kc+16+half*8 .. +7
            int n = (t << 4) + l15;
            v8bf blo = *(const v8bf*)(ldsW + (size_t)n * K + kc + (half << 3));
            v8bf bhi = *(const v8bf*)(ldsW + (size_t)n * K + kc + 16 + (half << 3));
            v16bf b = __builtin_shufflevector(blo, bhi, 0, 1, 2, 3, 4, 5, 6, 7,
                                              8, 9, 10, 11, 12, 13, 14, 15);
            acc[t] = __builtin_amdgcn_wmma_f32_16x16x32_bf16(
                false, a, false, b, (short)0, acc[t], false, false);
        }
    }
    // D layout: lane(0-15): N=lane, M=r ; lane(16-31): M=8+r
    #pragma unroll
    for (int t = 0; t < 8; ++t) {
        int n = (t << 4) + l15;
        #pragma unroll
        for (int r = 0; r < 8; ++r) {
            int row = m0 + (half << 3) + r;
            D[(size_t)row * HC_ + n] = acc[t][r];
        }
    }
}

// ---------- alpha_src / alpha_dst : [N,H] = sum_c h[n,h,c]*att[h,c] ----------
__global__ void alpha_kernel(const float* __restrict__ hfeat,
                             const float* __restrict__ att_src,
                             const float* __restrict__ att_dst,
                             float* __restrict__ as_, float* __restrict__ ad_)
{
    int t = blockIdx.x * blockDim.x + threadIdx.x;   // over N*H
    if (t >= Nn * Hh) return;
    int n = t >> 2, h = t & 3;
    const float4* row = (const float4*)(hfeat + (size_t)n * HC_ + h * Cc);
    const float4* s0  = (const float4*)(att_src + h * Cc);
    const float4* s1  = (const float4*)(att_dst + h * Cc);
    float a = 0.f, b = 0.f;
    #pragma unroll
    for (int c = 0; c < Cc / 4; ++c) {
        float4 r = row[c], u = s0[c], v = s1[c];
        a += r.x * u.x + r.y * u.y + r.z * u.z + r.w * u.w;
        b += r.x * v.x + r.y * v.y + r.z * v.z + r.w * v.w;
    }
    as_[t] = a; ad_[t] = b;
}

// ---------- edge pass 1: segment max (ordered-uint atomicMax) ----------
__global__ void edge_max_kernel(const int* __restrict__ ei,
                                const float* __restrict__ as_, const float* __restrict__ ad_,
                                unsigned* __restrict__ menc)
{
    int e = blockIdx.x * blockDim.x + threadIdx.x;
    if (e >= ETOT) return;
    int s, d;
    if (e < Ee) { s = ei[e]; d = ei[Ee + e]; } else { s = e - Ee; d = s; }
    #pragma unroll
    for (int h = 0; h < Hh; ++h) {
        float ev = as_[s * Hh + h] + ad_[d * Hh + h];
        ev = ev > 0.f ? ev : NEG_SLOPE_ * ev;
        atomicMax(&menc[d * Hh + h], fenc(ev));
    }
}

// ---------- edge pass 2: segment sum of exp(e - m) ----------
__global__ void edge_sum_kernel(const int* __restrict__ ei,
                                const float* __restrict__ as_, const float* __restrict__ ad_,
                                const unsigned* __restrict__ menc, float* __restrict__ z)
{
    int e = blockIdx.x * blockDim.x + threadIdx.x;
    if (e >= ETOT) return;
    int s, d;
    if (e < Ee) { s = ei[e]; d = ei[Ee + e]; } else { s = e - Ee; d = s; }
    #pragma unroll
    for (int h = 0; h < Hh; ++h) {
        float ev = as_[s * Hh + h] + ad_[d * Hh + h];
        ev = ev > 0.f ? ev : NEG_SLOPE_ * ev;
        atomicAdd(&z[d * Hh + h], expf(ev - fdec(menc[d * Hh + h])));
    }
}

// ---------- edge pass 3: out[dst] += w * h[src]; 1 wave/edge, float4 gathers ----------
__global__ __launch_bounds__(256) void edge_agg_kernel(
    const int* __restrict__ ei,
    const float* __restrict__ as_, const float* __restrict__ ad_,
    const unsigned* __restrict__ menc, const float* __restrict__ z,
    const float* __restrict__ hfeat, float* __restrict__ out)
{
    __shared__ float wsh[8][Hh];
    const int sub  = threadIdx.x >> 5;              // edge slot 0..7
    const int lane = threadIdx.x & 31;
    const long long e = (long long)blockIdx.x * 8 + sub;   // ETOT % 8 == 0
    int s, d;
    if (e < Ee) { s = ei[e]; d = ei[Ee + e]; } else { s = (int)(e - Ee); d = s; }
    if (lane < Hh) {
        int h = lane;
        float ev = as_[s * Hh + h] + ad_[d * Hh + h];
        ev = ev > 0.f ? ev : NEG_SLOPE_ * ev;
        float p = expf(ev - fdec(menc[d * Hh + h]));
        wsh[sub][h] = p / z[d * Hh + h];
    }
    __syncthreads();
    float w = wsh[sub][lane >> 3];                  // head = (lane*4)/32
    float4 hv = *(const float4*)(hfeat + (size_t)s * HC_ + lane * 4);
    float* o = out + (size_t)d * HC_ + lane * 4;
    atomicAdd(o + 0, w * hv.x);
    atomicAdd(o + 1, w * hv.y);
    atomicAdd(o + 2, w * hv.z);
    atomicAdd(o + 3, w * hv.w);
}

// ---------- bias + relu in place (float4) ----------
__global__ void bias_relu_kernel(float4* __restrict__ buf, const float4* __restrict__ b) {
    long long t = (long long)blockIdx.x * blockDim.x + threadIdx.x;   // over N*32
    if (t >= (long long)Nn * (HC_ / 4)) return;
    float4 v = buf[t], bb = b[t & (HC_ / 4 - 1)];
    v.x = fmaxf(v.x + bb.x, 0.f); v.y = fmaxf(v.y + bb.y, 0.f);
    v.z = fmaxf(v.z + bb.z, 0.f); v.w = fmaxf(v.w + bb.w, 0.f);
    buf[t] = v;
}

// ---------- global mean pool (accumulate, float4 reads) ----------
__global__ void pool_kernel(const float* __restrict__ hf, const int* __restrict__ batch,
                            float* __restrict__ sums, float* __restrict__ cnt)
{
    long long t = (long long)blockIdx.x * blockDim.x + threadIdx.x;   // over N*32
    if (t >= (long long)Nn * (HC_ / 4)) return;
    int n = (int)(t >> 5), q = (int)(t & 31);
    int g = batch[n];
    float4 v = *(const float4*)(hf + (size_t)n * HC_ + q * 4);
    float* sp = sums + g * HC_ + q * 4;
    atomicAdd(sp + 0, v.x); atomicAdd(sp + 1, v.y);
    atomicAdd(sp + 2, v.z); atomicAdd(sp + 3, v.w);
    if (q == 0) atomicAdd(&cnt[g], 1.0f);
}

// ---------- classifier: mean -> fc(128->32) -> fc(32->16) -> sigmoid ----------
__global__ __launch_bounds__(64) void classifier_kernel(
    const float* __restrict__ sums, const float* __restrict__ cnt,
    const float* __restrict__ Wc1, const float* __restrict__ bc1,
    const float* __restrict__ Wc2, const float* __restrict__ bc2,
    float* __restrict__ outp)
{
    int g = threadIdx.x;
    if (g >= NUM_GRAPHS_) return;
    float inv = 1.0f / cnt[g];
    float g1[Cc];
    #pragma unroll 4
    for (int i = 0; i < Cc; ++i) {
        float acc = bc1[i];
        for (int k = 0; k < HC_; ++k) acc += (sums[g * HC_ + k] * inv) * Wc1[k * Cc + i];
        g1[i] = acc;
    }
    for (int j = 0; j < NUM_CLASSES_; ++j) {
        float acc = bc2[j];
        #pragma unroll 8
        for (int i = 0; i < Cc; ++i) acc += g1[i] * Wc2[i * NUM_CLASSES_ + j];
        outp[g * NUM_CLASSES_ + j] = 1.0f / (1.0f + expf(-acc));
    }
}

// ---------- launch ----------
extern "C" void kernel_launch(void* const* d_in, const int* in_sizes, int n_in,
                              void* d_out, int out_size, void* d_ws, size_t ws_size,
                              hipStream_t stream) {
    const float* x        = (const float*)d_in[0];
    const int*   ei       = (const int*)d_in[1];
    const int*   batch    = (const int*)d_in[2];
    const float* W1       = (const float*)d_in[3];
    const float* att_src1 = (const float*)d_in[4];
    const float* att_dst1 = (const float*)d_in[5];
    const float* b1       = (const float*)d_in[6];
    const float* W2       = (const float*)d_in[7];
    const float* att_src2 = (const float*)d_in[8];
    const float* att_dst2 = (const float*)d_in[9];
    const float* b2       = (const float*)d_in[10];
    const float* Wc1      = (const float*)d_in[11];
    const float* bc1      = (const float*)d_in[12];
    const float* Wc2      = (const float*)d_in[13];
    const float* bc2      = (const float*)d_in[14];
    float* outp = (float*)d_out;

    // workspace layout
    float*    bufA = (float*)d_ws;                       // N*HC  (h = x@W)
    float*    bufB = bufA + (size_t)Nn * HC_;            // N*HC  (agg out / relu)
    float*    as_  = bufB + (size_t)Nn * HC_;            // N*H
    float*    ad_  = as_ + (size_t)Nn * Hh;              // N*H
    unsigned* menc = (unsigned*)(ad_ + (size_t)Nn * Hh); // N*H encoded max
    float*    z    = (float*)(menc + (size_t)Nn * Hh);   // N*H
    float*    sums = z + (size_t)Nn * Hh;                // G*HC
    float*    cnt  = sums + NUM_GRAPHS_ * HC_;           // G
    __bf16*   abf  = (__bf16*)(cnt + NUM_GRAPHS_);       // N*F_IN bf16 (reused layer2)
    __bf16*   wt   = abf + (size_t)Nn * F_IN_;           // 128*256 bf16 (transposed W)

    const long long nhc = (long long)Nn * HC_;
    const int gNHC4 = (int)((nhc / 4 + 255) / 256);
    const int gNH   = (Nn * Hh + 255) / 256;
    const int gNH4  = (Nn * Hh / 4 + 255) / 256;
    const int gEdge = (ETOT + 255) / 256;
    const int gAgg  = ETOT / 8;
    const int gGemm = (MTILES + 3) / 4;

    // ===== layer 1 =====
    cvt_bf16_kernel<<<(int)(((long long)Nn * F_IN_ / 4 + 255) / 256), 256, 0, stream>>>(
        (const float4*)x, (uint2*)abf, (long long)Nn * F_IN_ / 4);
    wtrans_kernel<<<(F_IN_ * HC_ + 255) / 256, 256, 0, stream>>>(W1, wt, F_IN_);
    gemm_wmma_bf16<<<gGemm, 128, 0, stream>>>(abf, wt, bufA, F_IN_);
    alpha_kernel<<<gNH, 256, 0, stream>>>(bufA, att_src1, att_dst1, as_, ad_);
    zero4_kernel<<<gNHC4, 256, 0, stream>>>((float4*)bufB, nhc / 4);
    zero4_kernel<<<gNH4, 256, 0, stream>>>((float4*)menc, Nn * Hh / 4);  // 0 == encoded -inf
    zero4_kernel<<<gNH4, 256, 0, stream>>>((float4*)z, Nn * Hh / 4);
    edge_max_kernel<<<gEdge, 256, 0, stream>>>(ei, as_, ad_, menc);
    edge_sum_kernel<<<gEdge, 256, 0, stream>>>(ei, as_, ad_, menc, z);
    edge_agg_kernel<<<gAgg, 256, 0, stream>>>(ei, as_, ad_, menc, z, bufA, bufB);
    bias_relu_kernel<<<gNHC4, 256, 0, stream>>>((float4*)bufB, (const float4*)b1);

    // ===== layer 2 ===== (bufB -> bf16 -> GEMM -> bufA, reuse bufB as output)
    cvt_bf16_kernel<<<gNHC4, 256, 0, stream>>>((const float4*)bufB, (uint2*)abf, nhc / 4);
    wtrans_kernel<<<(HC_ * HC_ + 255) / 256, 256, 0, stream>>>(W2, wt, HC_);
    gemm_wmma_bf16<<<gGemm, 128, 0, stream>>>(abf, wt, bufA, HC_);
    alpha_kernel<<<gNH, 256, 0, stream>>>(bufA, att_src2, att_dst2, as_, ad_);
    zero4_kernel<<<gNHC4, 256, 0, stream>>>((float4*)bufB, nhc / 4);
    zero4_kernel<<<gNH4, 256, 0, stream>>>((float4*)menc, Nn * Hh / 4);
    zero4_kernel<<<gNH4, 256, 0, stream>>>((float4*)z, Nn * Hh / 4);
    edge_max_kernel<<<gEdge, 256, 0, stream>>>(ei, as_, ad_, menc);
    edge_sum_kernel<<<gEdge, 256, 0, stream>>>(ei, as_, ad_, menc, z);
    edge_agg_kernel<<<gAgg, 256, 0, stream>>>(ei, as_, ad_, menc, z, bufA, bufB);
    bias_relu_kernel<<<gNHC4, 256, 0, stream>>>((float4*)bufB, (const float4*)b2);

    // ===== pool + classifier =====
    zero4_kernel<<<((NUM_GRAPHS_ * HC_ + NUM_GRAPHS_) / 4 + 255) / 256, 256, 0, stream>>>(
        (float4*)sums, (NUM_GRAPHS_ * HC_ + NUM_GRAPHS_) / 4);
    pool_kernel<<<gNHC4, 256, 0, stream>>>(bufB, batch, sums, cnt);
    classifier_kernel<<<1, 64, 0, stream>>>(sums, cnt, Wc1, bc1, Wc2, bc2, outp);
}
